// Attention_38852274160231
// MI455X (gfx1250) — matrix operand order repaired and checked by reading, not compiled, not run on hardware
//
#include <hip/hip_runtime.h>

typedef __attribute__((ext_vector_type(16))) __bf16 v16bf;
typedef __attribute__((ext_vector_type(8)))  float  v8f;

#define NHEADS 16
#define DQK    576
#define DVV    512
#define SEQLEN 1024
#define BN     64
#define QSTR   584            // 576 + 8 pad (bf16 elems)
#define VSTR   72             // 64 keys + 8 pad (bf16 elems)
#define PSTR   72
#define QLDS_N (16 * QSTR)    // 9344 ushorts
#define VLDS_N (DVV * VSTR)   // 36864 ushorts
#define PLDS_N (16 * PSTR)    // 1152 ushorts
#define SMEM_USHORTS (QLDS_N + VLDS_N + PLDS_N)       // 47360
#define SMEM_BYTES   (SMEM_USHORTS * 2 + 2 * 64 * 4)  // + redmax/redsum = 95232 B

struct __attribute__((aligned(16))) U4 { unsigned int x, y, z, w; };
struct __attribute__((aligned(8)))  U2 { unsigned int x, y; };
struct __attribute__((aligned(16))) F4 { float x, y, z, w; };

union Frag {
  struct { U4 lo, hi; } u;
  v16bf v;
};

// Hardware packed f32->bf16 (RNE) conversion: 1 VALU op for 2 elements.
__device__ __forceinline__ unsigned int bf16pack2(float lo, float hi) {
  unsigned int r;
  asm("v_cvt_pk_bf16_f32 %0, %1, %2" : "=v"(r) : "v"(lo), "v"(hi));
  return r;
}
__device__ __forceinline__ unsigned short f32_to_bf16(float f) {
  return (unsigned short)(bf16pack2(f, 0.0f) & 0xffffu);
}

__global__ __launch_bounds__(128, 1)
void mla_prefill_attn(const float* __restrict__ Q, const float* __restrict__ K,
                      const float* __restrict__ V, float* __restrict__ O) {
  extern __shared__ __align__(16) unsigned short smem[];
  unsigned short* Qlds = smem;
  unsigned short* Vlds = smem + QLDS_N;
  unsigned short* Plds = smem + QLDS_N + VLDS_N;
  float* redmax = (float*)(smem + SMEM_USHORTS);   // [4][16]
  float* redsum = redmax + 64;                     // [4][16]

  const int qt   = blockIdx.x;          // 16-query tile within segment
  const int h    = blockIdx.y;
  const int b    = blockIdx.z;
  const int tid  = threadIdx.x;
  const int wave = tid >> 5;
  const int lane = tid & 31;
  const int half = lane >> 4;           // 0: lanes 0-15, 1: lanes 16-31
  const int col  = lane & 15;

  const long seg0 = (long)b * SEQLEN;
  const long tok0 = seg0 + (long)qt * 16;

  // ---- stage Q tile (16 x 576) as bf16 into LDS (row-major, padded) ----
  for (int i = 0; i < 18; ++i) {
    const int e4  = tid + 128 * i;          // float4 index in [0, 2304)
    const int row = e4 / 144;
    const int d   = (e4 % 144) * 4;
    const F4 f = *(const F4*)(Q + ((tok0 + row) * NHEADS + h) * DQK + d);
    U2 pk;
    pk.x = bf16pack2(f.x, f.y);
    pk.y = bf16pack2(f.z, f.w);
    *(U2*)(&Qlds[row * QSTR + d]) = pk;
  }

  v8f o[8] = {};                          // O accumulators: 8 x (16x16 f32)
  float m[8], l[8];
  for (int r = 0; r < 8; ++r) { m[r] = -__builtin_inff(); l[r] = 0.0f; }

  const float kScaleLog2 = 1.44269504088896f / 24.0f;   // log2(e)/sqrt(576)

  const int niter = qt / 4 + 1;           // ceil((16*qt+16)/64)
  for (int it = 0; it < niter; ++it) {
    const int k0 = it * BN;
    __syncthreads();                      // prev iter done with Vlds/Plds/red*

    // ---- stage V tile (64 keys x 512) transposed bf16: Vlds[dv][key] ----
    {
      const int dv0 = tid * 4;            // each thread owns one 4-wide dv chunk
      const float* vb = V + ((seg0 + k0) * NHEADS + h) * DQK + dv0;
      for (int kp = 0; kp < 32; ++kp) {
        const F4 f0 = *(const F4*)(vb + (2 * kp) * (NHEADS * DQK));
        const F4 f1 = *(const F4*)(vb + (2 * kp + 1) * (NHEADS * DQK));
        *(unsigned int*)(&Vlds[(dv0 + 0) * VSTR + 2 * kp]) = bf16pack2(f0.x, f1.x);
        *(unsigned int*)(&Vlds[(dv0 + 1) * VSTR + 2 * kp]) = bf16pack2(f0.y, f1.y);
        *(unsigned int*)(&Vlds[(dv0 + 2) * VSTR + 2 * kp]) = bf16pack2(f0.z, f1.z);
        *(unsigned int*)(&Vlds[(dv0 + 3) * VSTR + 2 * kp]) = bf16pack2(f0.w, f1.w);
      }
    }

    // ---- Q·K^T: wave w computes S[0:16, 16w:16w+16], K-loop over 576 ----
    v8f cs = {};
    const int mykey = k0 + 16 * wave + col;                 // this lane's key column
    const float* kb = K + ((seg0 + mykey) * NHEADS + h) * DQK;
    if (it + 1 < niter)                   // prefetch next K tile row (global_prefetch_b8)
      __builtin_prefetch(kb + (long)BN * NHEADS * DQK, 0, 1);
    const int bkoff = half * 16;                            // B: contiguous 16 K-vals/lane
    for (int s = 0; s < 18; ++s) {
      const int d0 = 32 * s;
      Frag A, Bf;
      A.u.lo = *(const U4*)(&Qlds[col * QSTR + d0 + half * 16]);       // K 0-7 / 16-23
      A.u.hi = *(const U4*)(&Qlds[col * QSTR + d0 + half * 16 + 8]);   // K 8-15 / 24-31
      const F4 b0 = *(const F4*)(kb + d0 + bkoff);
      const F4 b1 = *(const F4*)(kb + d0 + bkoff + 4);
      const F4 b2 = *(const F4*)(kb + d0 + bkoff + 8);
      const F4 b3 = *(const F4*)(kb + d0 + bkoff + 12);
      Bf.u.lo.x = bf16pack2(b0.x, b0.y);  Bf.u.lo.y = bf16pack2(b0.z, b0.w);
      Bf.u.lo.z = bf16pack2(b1.x, b1.y);  Bf.u.lo.w = bf16pack2(b1.z, b1.w);
      Bf.u.hi.x = bf16pack2(b2.x, b2.y);  Bf.u.hi.y = bf16pack2(b2.z, b2.w);
      Bf.u.hi.z = bf16pack2(b3.x, b3.y);  Bf.u.hi.w = bf16pack2(b3.z, b3.w);
      cs = __builtin_amdgcn_wmma_f32_16x16x32_bf16(false, A.v, false, Bf.v,
                                                   (short)0, cs, false, false);
    }

    // ---- scale + causal mask (log2 domain); C layout: M = r + 8*half, N = col ----
    float sv[8];
    for (int r = 0; r < 8; ++r) {
      const int qrow = qt * 16 + r + 8 * half;
      sv[r] = (mykey <= qrow) ? cs[r] * kScaleLog2 : -__builtin_inff();
    }

    // ---- row max over this wave's 16 keys (half-wave shuffle reduce) ----
    float rmx[8];
    for (int r = 0; r < 8; ++r) rmx[r] = sv[r];
    for (int off = 1; off < 16; off <<= 1)
      for (int r = 0; r < 8; ++r)
        rmx[r] = fmaxf(rmx[r], __shfl_xor(rmx[r], off, 32));
    if (col == 0)
      for (int r = 0; r < 8; ++r)
        redmax[wave * 16 + r + 8 * half] = rmx[r];
    __syncthreads();                      // also covers V staging completion

    float alpha[8], p[8], rs[8];
    for (int r = 0; r < 8; ++r) {
      const int M = r + 8 * half;
      const float g = fmaxf(fmaxf(redmax[M], redmax[16 + M]),
                            fmaxf(redmax[32 + M], redmax[48 + M]));
      const float mn = fmaxf(m[r], g);
      alpha[r] = exp2f(m[r] - mn);
      p[r]     = exp2f(sv[r] - mn);       // masked -> exp2(-inf) = 0
      m[r]     = mn;
      rs[r]    = p[r];
    }
    for (int off = 1; off < 16; off <<= 1)
      for (int r = 0; r < 8; ++r)
        rs[r] += __shfl_xor(rs[r], off, 32);
    if (col == 0)
      for (int r = 0; r < 8; ++r)
        redsum[wave * 16 + r + 8 * half] = rs[r];
    for (int r = 0; r < 8; ++r)           // publish P tile (bf16)
      Plds[(r + 8 * half) * PSTR + 16 * wave + col] = f32_to_bf16(p[r]);
    __syncthreads();

    for (int r = 0; r < 8; ++r) {
      const int M = r + 8 * half;
      l[r] = l[r] * alpha[r] +
             (redsum[M] + redsum[16 + M] + redsum[32 + M] + redsum[48 + M]);
    }
    for (int s = 0; s < 8; ++s)           // rescale running output
      for (int r = 0; r < 8; ++r)
        o[s][r] *= alpha[r];

    // ---- P·V: wave w computes O[0:16, 128w : 128w+128], K-loop over 64 keys ----
    for (int kk = 0; kk < 2; ++kk) {
      Frag A;
      A.u.lo = *(const U4*)(&Plds[col * PSTR + kk * 32 + half * 16]);
      A.u.hi = *(const U4*)(&Plds[col * PSTR + kk * 32 + half * 16 + 8]);
      const int kbase = kk * 32 + half * 16;
      for (int s = 0; s < 8; ++s) {
        Frag Bv;
        const int dv = wave * 128 + s * 16 + col;
        Bv.u.lo = *(const U4*)(&Vlds[dv * VSTR + kbase]);
        Bv.u.hi = *(const U4*)(&Vlds[dv * VSTR + kbase + 8]);
        o[s] = __builtin_amdgcn_wmma_f32_16x16x32_bf16(false, A.v, false, Bv.v,
                                                       (short)0, o[s], false, false);
      }
    }
  }

  // ---- epilogue: O /= l, store fp32 [token][head][512] ----
  float invl[8];
  for (int r = 0; r < 8; ++r) invl[r] = 1.0f / l[r];
  for (int s = 0; s < 8; ++s)
    for (int r = 0; r < 8; ++r) {
      const int M = r + 8 * half;
      O[(tok0 + M) * (NHEADS * DVV) + h * DVV + wave * 128 + s * 16 + col] =
          o[s][r] * invl[r];
    }
}

extern "C" void kernel_launch(void* const* d_in, const int* in_sizes, int n_in,
                              void* d_out, int out_size, void* d_ws, size_t ws_size,
                              hipStream_t stream) {
  const float* q = (const float*)d_in[0];
  const float* k = (const float*)d_in[1];
  const float* v = (const float*)d_in[2];
  float* out = (float*)d_out;
  (void)in_sizes; (void)n_in; (void)out_size; (void)d_ws; (void)ws_size;

  (void)hipFuncSetAttribute((const void*)mla_prefill_attn,
                            hipFuncAttributeMaxDynamicSharedMemorySize, SMEM_BYTES);
  dim3 grid(SEQLEN / 16, NHEADS, 4);      // (q-tile, head, batch); x fastest -> L2 reuse
  mla_prefill_attn<<<grid, dim3(128, 1, 1), SMEM_BYTES, stream>>>(q, k, v, out);
}
